// LSTMModel_36077725286418
// MI455X (gfx1250) — matrix operand order repaired and checked by reading, not compiled, not run on hardware
//
#include <hip/hip_runtime.h>
#include <stdint.h>

// ---------------------------------------------------------------------------
// CDNA5 (gfx1250) stacked-LSTM inference, bf16 WMMA (V_WMMA_F32_16X16X32_BF16)
// Gate-aligned tiling: gates fused into the GEMM epilogue in registers.
// ---------------------------------------------------------------------------

typedef __bf16 bf16_t;
typedef bf16_t v16bf __attribute__((ext_vector_type(16)));
typedef float  v8f   __attribute__((ext_vector_type(8)));

#define NLSTM  8
#define NDENSE 5
#define BATCH  1024
#define SEQT   120
#define XDIM   132
#define XPAD   160   // pad K of layer-0 input projection to a multiple of 32

// fp32 -> bf16 round-to-nearest-even
__device__ __forceinline__ unsigned short f2bf(float x) {
  union { float f; unsigned u; } v; v.f = x;
  unsigned r = v.u + 0x7fffu + ((v.u >> 16) & 1u);
  return (unsigned short)(r >> 16);
}

union Frag { uint4 q[2]; v16bf v; };

// A fragment (16x32 bf16): lane half 0 -> K {0..7,16..23}; half 1 -> K {8..15,24..31}
__device__ __forceinline__ v16bf load_afrag(const unsigned short* p, int half) {
  Frag f;
  f.q[0] = *(const uint4*)(p + half * 8);
  f.q[1] = *(const uint4*)(p + half * 8 + 16);
  return f.v;
}
// B fragment (32x16 bf16, weights stored transposed [N][K]):
// lane half 0 -> K {0..7,16..23} ; half 1 -> K {8..15,24..31}
__device__ __forceinline__ v16bf load_bfrag(const unsigned short* p, int half) {
  Frag f;
  f.q[0] = *(const uint4*)(p + half * 8);
  f.q[1] = *(const uint4*)(p + half * 8 + 16);
  return f.v;
}

__device__ __forceinline__ float sigf(float x) { return 1.f / (1.f + __expf(-x)); }

#define WMMA_BF16(a, b, c) \
  __builtin_amdgcn_wmma_f32_16x16x32_bf16(false, (a), false, (b), (short)0, (c), false, false)

// ---------------------------------------------------------------------------
// Fully fused per-timestep LSTM kernel:
//   z = x_t @ Wk + h_{t-1} @ Wr + bias   (concatenated-K GEMM, bf16 WMMA)
//   c = sig(z_f)*c + sig(z_i)*relu(z_g);  h = sig(z_o)*relu(c)   (in registers)
// Wave owns 32 batch rows x 16 units, with 4 gate accumulators per M-tile at
// column blocks g*u + j0 — i/f/g/o of element (b,j) share lane+index.
// Block: 256 thr (8 waves) = 4 M-waves x 2 J-waves -> 128 rows x 32 units.
// ---------------------------------------------------------------------------
__global__ __launch_bounds__(256) void lstm_step_fused(
    const unsigned short* __restrict__ xseq, int t, int dpad, int xstride,
    const unsigned short* __restrict__ hprev,  // [B][u] bf16 (state t-1)
    unsigned short* __restrict__ hnext,        // [B][u] bf16 (state t)
    float* __restrict__ c,                     // [B][u] fp32, updated in place
    const unsigned short* __restrict__ WkT,    // [4u][dpad] bf16
    const unsigned short* __restrict__ WrT,    // [4u][u]    bf16
    const float* __restrict__ bias,            // [4u] fp32
    unsigned short* __restrict__ hseq,         // [B][T][u] bf16 out sequence
    int u)
{
  const int lane = threadIdx.x & 31;
  const int wave = threadIdx.x >> 5;
  const int m0   = blockIdx.y * 128 + (wave & 3) * 32;
  const int j0   = blockIdx.x * 32 + (wave >> 2) * 16;
  const int r15  = lane & 15;
  const int half = lane >> 4;

  v8f acc[4][2] = {};  // [gate][mtile]

  // Phase 1: input projection, K over dpad
  {
    const unsigned short* a0p = xseq + (size_t)(m0 + r15) * xstride + (size_t)t * dpad;
    const unsigned short* a1p = a0p + (size_t)16 * xstride;
    for (int k = 0; k < dpad; k += 32) {
      v16bf a0 = load_afrag(a0p + k, half);
      v16bf a1 = load_afrag(a1p + k, half);
#pragma unroll
      for (int g = 0; g < 4; ++g) {
        v16bf bfr = load_bfrag(WkT + (size_t)(g * u + j0 + r15) * dpad + k, half);
        acc[g][0] = WMMA_BF16(a0, bfr, acc[g][0]);
        acc[g][1] = WMMA_BF16(a1, bfr, acc[g][1]);
      }
    }
  }
  // Phase 2: recurrent projection, K over u
  {
    const unsigned short* a0p = hprev + (size_t)(m0 + r15) * u;
    const unsigned short* a1p = a0p + (size_t)16 * u;
    for (int k = 0; k < u; k += 32) {
      v16bf a0 = load_afrag(a0p + k, half);
      v16bf a1 = load_afrag(a1p + k, half);
#pragma unroll
      for (int g = 0; g < 4; ++g) {
        v16bf bfr = load_bfrag(WrT + (size_t)(g * u + j0 + r15) * u + k, half);
        acc[g][0] = WMMA_BF16(a0, bfr, acc[g][0]);
        acc[g][1] = WMMA_BF16(a1, bfr, acc[g][1]);
      }
    }
  }

  // Epilogue: bias + gates entirely in registers, update c / h
  const int j   = j0 + r15;
  const float bi = bias[0 * u + j];
  const float bf = bias[1 * u + j];
  const float bg = bias[2 * u + j];
  const float bo = bias[3 * u + j];
#pragma unroll
  for (int mi = 0; mi < 2; ++mi) {
#pragma unroll
    for (int r = 0; r < 8; ++r) {
      const int    mm   = m0 + mi * 16 + half * 8 + r;
      const size_t sidx = (size_t)mm * u + j;
      const float ig = sigf(acc[0][mi][r] + bi);
      const float fg = sigf(acc[1][mi][r] + bf);
      const float gg = fmaxf(acc[2][mi][r] + bg, 0.f);
      const float og = sigf(acc[3][mi][r] + bo);
      const float cn = fg * c[sidx] + ig * gg;
      c[sidx] = cn;
      const float h = og * fmaxf(cn, 0.f);
      const unsigned short hb = f2bf(h);
      hnext[sidx] = hb;
      hseq[(size_t)mm * (SEQT * u) + (size_t)t * u + j] = hb;
    }
  }
}

// ---------------------------------------------------------------------------
// Generic WMMA GEMM: C = act(A @ B^T + bias); A bf16 [M][K] (lda), BT bf16 [Npad][K]
// Block tile 128x128, wave tile 32x64. Used for the dense head.
// ---------------------------------------------------------------------------
__global__ __launch_bounds__(256) void gemm_bias_act(
    const unsigned short* __restrict__ A, int lda,
    const unsigned short* __restrict__ BT, int ldb,
    const float* __restrict__ bias,
    float* __restrict__ Cf, unsigned short* __restrict__ Cbf,
    int N, int K, int ldc, int relu)
{
  const int lane = threadIdx.x & 31;
  const int wave = threadIdx.x >> 5;
  const int m0   = blockIdx.y * 128 + (wave & 3) * 32;
  const int n0   = blockIdx.x * 128 + (wave >> 2) * 64;
  const int r15  = lane & 15;
  const int half = lane >> 4;

  v8f acc[2][4] = {};

  const unsigned short* a0p = A + (size_t)(m0 + r15) * lda;
  const unsigned short* a1p = a0p + (size_t)16 * lda;
  for (int k = 0; k < K; k += 32) {
    v16bf a0 = load_afrag(a0p + k, half);
    v16bf a1 = load_afrag(a1p + k, half);
#pragma unroll
    for (int ni = 0; ni < 4; ++ni) {
      v16bf bfr = load_bfrag(BT + (size_t)(n0 + ni * 16 + r15) * ldb + k, half);
      acc[0][ni] = WMMA_BF16(a0, bfr, acc[0][ni]);
      acc[1][ni] = WMMA_BF16(a1, bfr, acc[1][ni]);
    }
  }
#pragma unroll
  for (int mi = 0; mi < 2; ++mi)
#pragma unroll
    for (int ni = 0; ni < 4; ++ni) {
      const int nn = n0 + ni * 16 + r15;
      if (nn < N) {
        const float bv = bias[nn];
#pragma unroll
        for (int r = 0; r < 8; ++r) {
          const int mm = m0 + mi * 16 + half * 8 + r;
          float v = acc[mi][ni][r] + bv;
          if (relu) v = fmaxf(v, 0.f);
          if (Cf)  Cf[(size_t)mm * ldc + nn]  = v;
          if (Cbf) Cbf[(size_t)mm * ldc + nn] = f2bf(v);
        }
      }
    }
}

__global__ void zero_state(float* __restrict__ c, unsigned short* __restrict__ h, int n) {
  const int idx = blockIdx.x * blockDim.x + threadIdx.x;
  if (idx >= n) return;
  c[idx] = 0.f; h[idx] = 0;
}

// Convert + zero-pad x: fp32 [B,T,132] -> bf16 [B,T,160]
__global__ void conv_pad_x(const float* __restrict__ x, unsigned short* __restrict__ out, int n) {
  const int idx = blockIdx.x * blockDim.x + threadIdx.x;
  if (idx >= n) return;
  const int d  = idx % XPAD;
  const int bt = idx / XPAD;
  out[idx] = (d < XDIM) ? f2bf(x[(size_t)bt * XDIM + d]) : (unsigned short)0;
}

// Convert + transpose + zero-pad weights: fp32 [K][N] -> bf16 [Npad][Kpad]
__global__ void conv_transpose(const float* __restrict__ in, unsigned short* __restrict__ out,
                               int K, int N, int Kpad, int Npad) {
  const int idx = blockIdx.x * blockDim.x + threadIdx.x;
  if (idx >= Kpad * Npad) return;
  const int k = idx % Kpad, nn = idx / Kpad;
  out[idx] = (k < K && nn < N) ? f2bf(in[(size_t)k * N + nn]) : (unsigned short)0;
}

__global__ void softmax10(const float* __restrict__ logits, float* __restrict__ out) {
  const int b = blockIdx.x * blockDim.x + threadIdx.x;  // 0..1023
  float v[10];
  float m = -3.4e38f;
#pragma unroll
  for (int i = 0; i < 10; ++i) { v[i] = logits[b * 10 + i]; m = fmaxf(m, v[i]); }
  float s = 0.f;
#pragma unroll
  for (int i = 0; i < 10; ++i) { v[i] = __expf(v[i] - m); s += v[i]; }
  const float inv = 1.f / s;
#pragma unroll
  for (int i = 0; i < 10; ++i) out[b * 10 + i] = v[i] * inv;
}

// ---------------------------------------------------------------------------
// Host orchestration
// ---------------------------------------------------------------------------
extern "C" void kernel_launch(void* const* d_in, const int* in_sizes, int n_in,
                              void* d_out, int out_size, void* d_ws, size_t ws_size,
                              hipStream_t stream) {
  static const int U[NLSTM]    = {128, 128, 256, 256, 512, 512, 256, 256};
  static const int DU[NDENSE]  = {256, 256, 128, 128, 10};
  static const int DUP[NDENSE] = {256, 256, 128, 128, 128};  // N padded to tile

  const float* x = (const float*)d_in[0];

  char* ws = (char*)d_ws;
  size_t off = 0;
  auto carve = [&](size_t bytes) -> void* {
    void* p = ws + off;
    off += (bytes + 255) & ~(size_t)255;
    return p;
  };

  const size_t SEQ_ELEMS = (size_t)BATCH * SEQT * 512;  // largest layer sequence
  unsigned short* seq[2];
  seq[0] = (unsigned short*)carve(SEQ_ELEMS * 2);
  seq[1] = (unsigned short*)carve(SEQ_ELEMS * 2);
  unsigned short* hp[2];
  hp[0] = (unsigned short*)carve((size_t)BATCH * 512 * 2);  // h state ping
  hp[1] = (unsigned short*)carve((size_t)BATCH * 512 * 2);  // h state pong
  float* cbuf = (float*)carve((size_t)BATCH * 512 * 4);
  unsigned short *lwk[NLSTM], *lwr[NLSTM], *dwt[NDENSE];
  for (int li = 0; li < NLSTM; ++li) {
    const int u = U[li];
    const int dpad = (li == 0) ? XPAD : U[li - 1];
    lwk[li] = (unsigned short*)carve((size_t)4 * u * dpad * 2);
    lwr[li] = (unsigned short*)carve((size_t)4 * u * u * 2);
  }
  int dk[NDENSE];
  int kd = 256;  // lstm7 units
  for (int di = 0; di < NDENSE; ++di) {
    dk[di]  = kd;
    dwt[di] = (unsigned short*)carve((size_t)DUP[di] * kd * 2);
    kd = DU[di];
  }
  float*          logits = (float*)carve((size_t)BATCH * 10 * 4);
  unsigned short* act[2];
  act[0] = (unsigned short*)carve((size_t)BATCH * 256 * 2);
  act[1] = (unsigned short*)carve((size_t)BATCH * 256 * 2);

  // --- weight prep: fp32 [K][N] -> bf16 [Npad][Kpad] (transposed, padded) ---
  for (int li = 0; li < NLSTM; ++li) {
    const int u = U[li];
    const int din  = (li == 0) ? XDIM : U[li - 1];
    const int dpad = (li == 0) ? XPAD : din;
    const float* Wk = (const float*)d_in[1 + 3 * li + 0];
    const float* Wr = (const float*)d_in[1 + 3 * li + 1];
    const int n1 = 4 * u * dpad;
    conv_transpose<<<(n1 + 255) / 256, 256, 0, stream>>>(Wk, lwk[li], din, 4 * u, dpad, 4 * u);
    const int n2 = 4 * u * u;
    conv_transpose<<<(n2 + 255) / 256, 256, 0, stream>>>(Wr, lwr[li], u, 4 * u, u, 4 * u);
  }
  for (int di = 0; di < NDENSE; ++di) {
    const float* Wd = (const float*)d_in[1 + 3 * NLSTM + 2 * di + 0];
    const int n = DUP[di] * dk[di];
    conv_transpose<<<(n + 255) / 256, 256, 0, stream>>>(Wd, dwt[di], dk[di], DU[di], dk[di], DUP[di]);
  }
  // --- input conversion ---
  {
    const int n = BATCH * SEQT * XPAD;
    conv_pad_x<<<(n + 255) / 256, 256, 0, stream>>>(x, seq[0], n);
  }

  // --- LSTM stack: one fused kernel per timestep, h state ping-pongs ---
  int cur = 0, dcur = XPAD;
  for (int li = 0; li < NLSTM; ++li) {
    const int u = U[li];
    const float* b = (const float*)d_in[1 + 3 * li + 2];
    const int ns = BATCH * u;
    zero_state<<<(ns + 255) / 256, 256, 0, stream>>>(cbuf, hp[0], ns);
    dim3 g(u / 32, BATCH / 128);
    for (int t = 0; t < SEQT; ++t) {
      lstm_step_fused<<<g, 256, 0, stream>>>(seq[cur], t, dcur, SEQT * dcur,
                                             hp[t & 1], hp[(t + 1) & 1], cbuf,
                                             lwk[li], lwr[li], b, seq[1 - cur], u);
    }
    cur = 1 - cur;
    dcur = u;
  }

  // --- Dense head (input = last timestep of final layer, u=256) ---
  const unsigned short* Ad = seq[cur] + (size_t)(SEQT - 1) * 256;
  int lda = SEQT * 256;
  for (int di = 0; di < NDENSE; ++di) {
    const float* bD = (const float*)d_in[1 + 3 * NLSTM + 2 * di + 1];
    dim3 g((DU[di] + 127) / 128, BATCH / 128);
    if (di < NDENSE - 1) {
      unsigned short* outb = act[di & 1];
      gemm_bias_act<<<g, 256, 0, stream>>>(Ad, lda, dwt[di], dk[di], bD,
                                           nullptr, outb, DU[di], dk[di], DU[di], 1);
      Ad = outb; lda = DU[di];
    } else {
      gemm_bias_act<<<g, 256, 0, stream>>>(Ad, lda, dwt[di], dk[di], bD,
                                           logits, nullptr, DU[di], dk[di], DU[di], 0);
    }
  }
  softmax10<<<BATCH / 256, 256, 0, stream>>>(logits, (float*)d_out);

  (void)in_sizes; (void)n_in; (void)out_size; (void)ws_size;
}
